// TriangleMultiplication_16999480558055
// MI455X (gfx1250) — matrix operand order repaired and checked by reading, not compiled, not run on hardware
//
#include <hip/hip_runtime.h>

// ---------------------------------------------------------------------------
// TriangleMultiplication on MI455X (gfx1250), wave32 + WMMA bf16.
//   L = 512, DZ = DC = 128, B = 1.
// Pipeline:
//   prep_kernel : LN(z1/z2/z3) -> 9 gated 128x128 projections (bf16 wmma),
//                 writes z12,z13,p2,p3 as [c][i][k] bf16 and gate5 [pos][c].
//   tri_kernel  : per-channel 512^3 GEMMs for both triangle terms, bf16 wmma,
//                 64x64 output blocks, K staged through LDS in 32-chunks.
//                 Tile staging uses GLOBAL_LOAD_ASYNC_TO_LDS when available.
//   final_kernel: LN5 + l5 GEMM + gate + ELU, f32 output in (C,L,L).
// ---------------------------------------------------------------------------

#define LL   262144   // 512*512
#define LDIM 512

typedef __bf16 bf16;
typedef __attribute__((ext_vector_type(8)))  bf16  v8bf;
typedef __attribute__((ext_vector_type(16))) bf16  v16bf;
typedef __attribute__((ext_vector_type(8)))  float v8f;

#ifndef __has_builtin
#define __has_builtin(x) 0
#endif
#if __has_builtin(__builtin_amdgcn_global_load_async_to_lds_b128) && \
    __has_builtin(__builtin_amdgcn_s_wait_asynccnt)
#define USE_ASYNC_LDS 1
#else
#define USE_ASYNC_LDS 0
#endif
#if USE_ASYNC_LDS && __has_builtin(__builtin_amdgcn_global_store_async_from_lds_b128)
#define USE_ASYNC_ST 1
#else
#define USE_ASYNC_ST 0
#endif

#if USE_ASYNC_LDS
// The builtins take generic pointers to a 4 x i32 *vector* type (HIP's uint4
// is a struct, so define a real vector type and cast through it).
typedef __attribute__((__vector_size__(4 * sizeof(int)))) int v4i_async;
static __device__ __forceinline__ void async_g2l_b128(const void* g, void* l) {
  __builtin_amdgcn_global_load_async_to_lds_b128((v4i_async*)g, (v4i_async*)l,
                                                 0, 0);
}
#endif
#if USE_ASYNC_ST
static __device__ __forceinline__ void async_l2g_b128(void* g, const void* l) {
  __builtin_amdgcn_global_store_async_from_lds_b128((v4i_async*)g,
                                                    (v4i_async*)l, 0, 0);
}
#endif

static __device__ __forceinline__ v8f wmma_bf16f32(v16bf a, v16bf b, v8f c) {
  // 8 args: (neg_a, A, neg_b, B, c_mod, C, reuse_a, reuse_b)
  return __builtin_amdgcn_wmma_f32_16x16x32_bf16(false, a, false, b, (short)0, c,
                                                 false, false);
}

// A-matrix (16x32 bf16) fragment from LDS [m][k] with leading dim ld.
// Lanes 0-15: row m0+lane, K = {k0..k0+7, k0+16..k0+23}
// Lanes 16-31: row m0+lane-16, K = {k0+8..k0+15, k0+24..k0+31}
static __device__ __forceinline__ v16bf frag_a(const bf16* sm, int ld, int m0,
                                               int k0, int lane) {
  int half = lane >> 4;
  int m = m0 + (lane & 15);
  const bf16* p = sm + m * ld + k0 + half * 8;
  v8bf lo = *(const v8bf*)p;
  v8bf hi = *(const v8bf*)(p + 16);
  return __builtin_shufflevector(lo, hi, 0, 1, 2, 3, 4, 5, 6, 7, 8, 9, 10, 11,
                                 12, 13, 14, 15);
}

// B-matrix (32x16 bf16) fragment from LDS stored column-major [n][k], ld = row len.
// Lanes 0-15: column n0+lane, K = {k0..k0+7, k0+16..k0+23} ; lanes 16-31: K = {k0+8..k0+15, k0+24..k0+31}.
static __device__ __forceinline__ v16bf frag_b(const bf16* sm, int ld, int n0,
                                               int k0, int lane) {
  int kh = lane >> 4;
  int n = n0 + (lane & 15);
  const bf16* p = sm + n * ld + k0 + kh * 8;
  v8bf lo = *(const v8bf*)p;
  v8bf hi = *(const v8bf*)(p + 16);
  return __builtin_shufflevector(lo, hi, 0, 1, 2, 3, 4, 5, 6, 7, 8, 9, 10, 11,
                                 12, 13, 14, 15);
}

static __device__ __forceinline__ float sigmoidf(float x) {
  return 1.0f / (1.0f + __expf(-x));
}

// --------------------- shared GEMM helpers (64 x 128 x 128) -----------------
// smA: 64x136 bf16 [pos][ch], smW: 128x136 bf16 [out][in].
// 8 waves, wave w: rows m0=(w>>1)*16, n-tile group (w&1)*4 .. +3.
static __device__ __forceinline__ void gemm_64x128(const bf16* smA,
                                                   const bf16* smW, v8f acc[4],
                                                   int wave, int lane) {
  int m0 = (wave >> 1) * 16;
  int ng = (wave & 1) * 4;
#pragma unroll
  for (int ks = 0; ks < 4; ++ks) {
    v16bf a = frag_a(smA, 136, m0, ks * 32, lane);
#pragma unroll
    for (int t = 0; t < 4; ++t) {
      v16bf b = frag_b(smW, 136, (ng + t) * 16, ks * 32, lane);
      acc[t] = wmma_bf16f32(a, b, acc[t]);
    }
  }
}

// Stage a 128x128 f32 weight (layout [in k][out n]) -> LDS bf16 [n][k], ld 136.
static __device__ __forceinline__ void stage_weight(const float* W, bf16* smW,
                                                    int tid) {
#pragma unroll 8
  for (int i = 0; i < 64; ++i) {
    int e = i * 256 + tid;
    int k = e >> 7;
    int n = e & 127;
    smW[n * 136 + k] = (bf16)W[e];
  }
}

// LayerNorm a 64-position tile of a (C,L,L) f32 tensor into smA bf16 [pos][ch].
static __device__ void ln_tile_f32(const float* z, const float* w,
                                   const float* b, int pos0, bf16* smA,
                                   float* red, float* red2, float* mean,
                                   float* rstd, int tid) {
  int p = tid & 63;
  int q = tid >> 6;  // 4 threads per position, 32 channels each
  const float* src = z + (size_t)(q * 32) * LL + pos0 + p;
  float s = 0.f, s2 = 0.f;
#pragma unroll 8
  for (int cc = 0; cc < 32; ++cc) {
    float v = src[(size_t)cc * LL];
    s += v;
    s2 += v * v;
  }
  red[tid] = s;
  red2[tid] = s2;
  __syncthreads();
  if (q == 0) {
    float S = red[p] + red[p + 64] + red[p + 128] + red[p + 192];
    float Q = red2[p] + red2[p + 64] + red2[p + 128] + red2[p + 192];
    float mu = S * (1.f / 128.f);
    float var = Q * (1.f / 128.f) - mu * mu;
    mean[p] = mu;
    rstd[p] = rsqrtf(var + 1e-5f);
  }
  __syncthreads();
  float mu = mean[p], rs = rstd[p];
#pragma unroll 8
  for (int cc = 0; cc < 32; ++cc) {
    int c = q * 32 + cc;
    float v = src[(size_t)cc * LL];
    smA[p * 136 + c] = (bf16)((v - mu) * rs * w[c] + b[c]);
  }
}

// Same for a bf16 source tensor (used by final_kernel).
static __device__ void ln_tile_bf16(const bf16* z, const float* w,
                                    const float* b, int pos0, bf16* smA,
                                    float* red, float* red2, float* mean,
                                    float* rstd, int tid) {
  int p = tid & 63;
  int q = tid >> 6;
  const bf16* src = z + (size_t)(q * 32) * LL + pos0 + p;
  float s = 0.f, s2 = 0.f;
#pragma unroll 8
  for (int cc = 0; cc < 32; ++cc) {
    float v = (float)src[(size_t)cc * LL];
    s += v;
    s2 += v * v;
  }
  red[tid] = s;
  red2[tid] = s2;
  __syncthreads();
  if (q == 0) {
    float S = red[p] + red[p + 64] + red[p + 128] + red[p + 192];
    float Q = red2[p] + red2[p + 64] + red2[p + 128] + red2[p + 192];
    float mu = S * (1.f / 128.f);
    float var = Q * (1.f / 128.f) - mu * mu;
    mean[p] = mu;
    rstd[p] = rsqrtf(var + 1e-5f);
  }
  __syncthreads();
  float mu = mean[p], rs = rstd[p];
#pragma unroll 8
  for (int cc = 0; cc < 32; ++cc) {
    int c = q * 32 + cc;
    float v = (float)src[(size_t)cc * LL];
    smA[p * 136 + c] = (bf16)((v - mu) * rs * w[c] + b[c]);
  }
}

// gated projection: dst[c][pos] = (A@WL + bL) * sigmoid(A@WG + bG), bf16 out,
// channel-major global layout. smW region doubles as the 128x64 out-staging.
static __device__ void gated_proj(const bf16* smA, bf16* smW, const float* wL,
                                  const float* bL, const float* wG,
                                  const float* bG, bf16* dst, int pos0,
                                  int tid) {
  int wave = tid >> 5, lane = tid & 31;
  stage_weight(wL, smW, tid);
  __syncthreads();
  v8f accL[4] = {{}, {}, {}, {}};
  gemm_64x128(smA, smW, accL, wave, lane);
  __syncthreads();
  stage_weight(wG, smW, tid);
  __syncthreads();
  v8f accG[4] = {{}, {}, {}, {}};
  gemm_64x128(smA, smW, accG, wave, lane);
  __syncthreads();
  bf16* smOut = smW;  // reuse region: [ch][pos], ld 64
  int ng = (wave & 1) * 4, m0 = (wave >> 1) * 16;
  int nl = lane & 15, mbase = (lane >> 4) * 8;
#pragma unroll
  for (int t = 0; t < 4; ++t) {
    int n = (ng + t) * 16 + nl;
    float biasL = bL[n], biasG = bG[n];
#pragma unroll
    for (int r = 0; r < 8; ++r) {
      int m = m0 + mbase + r;
      float v = accL[t][r] + biasL;
      float g = sigmoidf(accG[t][r] + biasG);
      smOut[n * 64 + m] = (bf16)(v * g);
    }
  }
  __syncthreads();
  {  // coalesced store: 128 channel rows x 64 contiguous positions
    int row = tid >> 1, half = tid & 1;
    const uint4* s = (const uint4*)(smOut + row * 64 + half * 32);
    uint4* d = (uint4*)(dst + (size_t)row * LL + pos0 + half * 32);
    d[0] = s[0]; d[1] = s[1]; d[2] = s[2]; d[3] = s[3];
  }
  __syncthreads();
}

// gate-only projection: dst[pos][ch] = sigmoid(A@W + b), bf16.
static __device__ void gate_proj(const bf16* smA, bf16* smW, const float* wG,
                                 const float* bG, bf16* dst, int pos0,
                                 int tid) {
  int wave = tid >> 5, lane = tid & 31;
  stage_weight(wG, smW, tid);
  __syncthreads();
  v8f acc[4] = {{}, {}, {}, {}};
  gemm_64x128(smA, smW, acc, wave, lane);
  __syncthreads();
  bf16* smOut = smW;  // reuse region: [pos][ch], ld 128 (fully contiguous tile)
  int ng = (wave & 1) * 4, m0 = (wave >> 1) * 16;
  int nl = lane & 15, mbase = (lane >> 4) * 8;
#pragma unroll
  for (int t = 0; t < 4; ++t) {
    int n = (ng + t) * 16 + nl;
    float bias = bG[n];
#pragma unroll
    for (int r = 0; r < 8; ++r) {
      int m = m0 + mbase + r;
      smOut[m * 128 + n] = (bf16)sigmoidf(acc[t][r] + bias);
    }
  }
  __syncthreads();
  {  // tile is contiguous in global: 1024 uint4
    const uint4* s = (const uint4*)smOut;
    uint4* d = (uint4*)(dst + (size_t)pos0 * 128);
    d[tid] = s[tid];
    d[tid + 256] = s[tid + 256];
    d[tid + 512] = s[tid + 512];
    d[tid + 768] = s[tid + 768];
  }
  __syncthreads();
}

// ------------------------------- kernel 1 ----------------------------------
__global__ __launch_bounds__(256) void prep_kernel(
    const float* z1, const float* z2, const float* z3, const float* ln1w,
    const float* ln1b, const float* ln2w, const float* ln2b, const float* ln3w,
    const float* ln3b, const float* l1w, const float* l1b, const float* g1w,
    const float* g1b, const float* l2w, const float* l2b, const float* g2w,
    const float* g2b, const float* l3w, const float* l3b, const float* g3w,
    const float* g3b, const float* l4w, const float* l4b, const float* g4w,
    const float* g4b, const float* g5w, const float* g5b, bf16* z12, bf16* z13,
    bf16* p2o, bf16* p3o, bf16* gate5) {
  __shared__ alignas(32) char smem[54784];
  bf16* smA = (bf16*)smem;                   // 64*136*2  = 17408
  bf16* smW = (bf16*)(smem + 17408);         // 128*136*2 = 34816 (also out-staging)
  float* red = (float*)(smem + 52224);
  float* red2 = (float*)(smem + 53248);
  float* mean = (float*)(smem + 54272);
  float* rstd = (float*)(smem + 54528);
  int tid = threadIdx.x;
  int pos0 = blockIdx.x * 64;

  // ---- src a = LN(z1): l1/g1 -> z12, l2/g2 -> z13, g5 -> gate5 ----
  ln_tile_f32(z1, ln1w, ln1b, pos0, smA, red, red2, mean, rstd, tid);
  __syncthreads();
  gated_proj(smA, smW, l1w, l1b, g1w, g1b, z12, pos0, tid);
  gated_proj(smA, smW, l2w, l2b, g2w, g2b, z13, pos0, tid);
  gate_proj(smA, smW, g5w, g5b, gate5, pos0, tid);

  // ---- src b = LN(z2): l3/g3 -> p2 ----
  ln_tile_f32(z2, ln2w, ln2b, pos0, smA, red, red2, mean, rstd, tid);
  __syncthreads();
  gated_proj(smA, smW, l3w, l3b, g3w, g3b, p2o, pos0, tid);

  // ---- src c = LN(z3): l4/g4 -> p3 ----
  ln_tile_f32(z3, ln3w, ln3b, pos0, smA, red, red2, mean, rstd, tid);
  __syncthreads();
  gated_proj(smA, smW, l4w, l4b, g4w, g4b, p3o, pos0, tid);
}

// ------------------------------- kernel 2 ----------------------------------
// out[c][i][j] = sum_k p2[c][i][k]*z12[c][k][j] + sum_k z13[c][i][k]*p3[c][j][k]
__global__ __launch_bounds__(256) void tri_kernel(const bf16* z12,
                                                  const bf16* z13,
                                                  const bf16* p2,
                                                  const bf16* p3, bf16* zsum) {
  __shared__ alignas(32) char smem[28672];
  bf16* sA1 = (bf16*)smem;             // p2  tile [i][k]  64x40
  bf16* sB1 = (bf16*)(smem + 5120);    // z12 tile transposed -> [j][k]
  bf16* sA2 = (bf16*)(smem + 10240);   // z13 tile [i][k]
  bf16* sB2 = (bf16*)(smem + 15360);   // p3  tile [j][k] (natural layout)
  bf16* sOut = (bf16*)(smem + 20480);  // 64x64 output staging
  int tid = threadIdx.x;
  int wave = tid >> 5, lane = tid & 31;
  int c = blockIdx.y;
  int i0 = (blockIdx.x >> 3) * 64;
  int j0 = (blockIdx.x & 7) * 64;
  const bf16* p2c = p2 + (size_t)c * LL;
  const bf16* z12c = z12 + (size_t)c * LL;
  const bf16* z13c = z13 + (size_t)c * LL;
  const bf16* p3c = p3 + (size_t)c * LL;

  v8f acc0 = {}, acc1 = {};
  int m0 = (wave >> 1) * 16;
  int n0a = ((wave & 1) * 2) * 16;
  int n0b = n0a + 16;
  int nl = lane & 15, mbase = (lane >> 4) * 8;

  for (int ks = 0; ks < 16; ++ks) {
    int k0 = ks * 32;
    {  // A-style tiles: 64 rows x 32 bf16 (64B/row), plus p3 natural [j][k].
      int row = tid >> 2, seg = tid & 3;
#if USE_ASYNC_LDS
      // CDNA5 async DMA: global -> LDS, 16B per lane, no VGPR staging.
      async_g2l_b128(p2c + (size_t)(i0 + row) * LDIM + k0 + seg * 8,
                     sA1 + row * 40 + seg * 8);
      async_g2l_b128(z13c + (size_t)(i0 + row) * LDIM + k0 + seg * 8,
                     sA2 + row * 40 + seg * 8);
      async_g2l_b128(p3c + (size_t)(j0 + row) * LDIM + k0 + seg * 8,
                     sB2 + row * 40 + seg * 8);
#else
      *(uint4*)(sA1 + row * 40 + seg * 8) =
          *((const uint4*)(p2c + (size_t)(i0 + row) * LDIM + k0) + seg);
      *(uint4*)(sA2 + row * 40 + seg * 8) =
          *((const uint4*)(z13c + (size_t)(i0 + row) * LDIM + k0) + seg);
      *(uint4*)(sB2 + row * 40 + seg * 8) =
          *((const uint4*)(p3c + (size_t)(j0 + row) * LDIM + k0) + seg);
#endif
    }
    {  // z12 [k][j] -> LDS transposed [j][k] (needs element transpose)
      int k = tid >> 3;
      int jj = (tid & 7) * 8;
      v8bf v = *(const v8bf*)(z12c + (size_t)(k0 + k) * LDIM + j0 + jj);
#pragma unroll
      for (int u = 0; u < 8; ++u) sB1[(jj + u) * 40 + k] = v[u];
    }
#if USE_ASYNC_LDS
    __builtin_amdgcn_s_wait_asynccnt(0);
#endif
    __syncthreads();
    v16bf a1 = frag_a(sA1, 40, m0, 0, lane);
    v16bf a2 = frag_a(sA2, 40, m0, 0, lane);
    v16bf b1a = frag_b(sB1, 40, n0a, 0, lane);
    v16bf b2a = frag_b(sB2, 40, n0a, 0, lane);
    acc0 = wmma_bf16f32(a1, b1a, acc0);
    acc0 = wmma_bf16f32(a2, b2a, acc0);
    v16bf b1b = frag_b(sB1, 40, n0b, 0, lane);
    v16bf b2b = frag_b(sB2, 40, n0b, 0, lane);
    acc1 = wmma_bf16f32(a1, b1b, acc1);
    acc1 = wmma_bf16f32(a2, b2b, acc1);
    if (ks < 15) {  // hint next k-stage into cache (global_prefetch_b8)
      __builtin_prefetch(p2c + (size_t)(i0 + (tid & 63)) * LDIM + k0 + 32, 0, 1);
      __builtin_prefetch(z13c + (size_t)(i0 + (tid & 63)) * LDIM + k0 + 32, 0, 1);
      __builtin_prefetch(p3c + (size_t)(j0 + (tid & 63)) * LDIM + k0 + 32, 0, 1);
      __builtin_prefetch(z12c + (size_t)(k0 + 32 + (tid & 31)) * LDIM + j0, 0, 1);
    }
    __syncthreads();
  }

#pragma unroll
  for (int r = 0; r < 8; ++r) {
    int m = m0 + mbase + r;
    sOut[m * 64 + n0a + nl] = (bf16)acc0[r];
    sOut[m * 64 + n0b + nl] = (bf16)acc1[r];
  }
  __syncthreads();
  {  // 64 rows x 128B coalesced store
    int row = tid >> 2, seg = tid & 3;
    bf16* gdst = zsum + (size_t)c * LL + (size_t)(i0 + row) * LDIM + j0 + seg * 16;
#if USE_ASYNC_ST
    // async LDS -> global store; s_wait_asynccnt(0) below drains it.
    async_l2g_b128(gdst, sOut + row * 64 + seg * 16);
    async_l2g_b128(gdst + 8, sOut + row * 64 + seg * 16 + 8);
    __builtin_amdgcn_s_wait_asynccnt(0);
#else
    const uint4* s = (const uint4*)(sOut + row * 64 + seg * 16);
    uint4* d = (uint4*)gdst;
    d[0] = s[0];
    d[1] = s[1];
#endif
  }
}

// ------------------------------- kernel 3 ----------------------------------
__global__ __launch_bounds__(256) void final_kernel(
    const bf16* zsum, const bf16* gate5, const float* ln5w, const float* ln5b,
    const float* l5w, const float* l5b, float* out) {
  __shared__ alignas(32) char smem[54784];
  bf16* smA = (bf16*)smem;
  bf16* smW = (bf16*)(smem + 17408);  // also reused as f32 out-staging (32KB)
  float* red = (float*)(smem + 52224);
  float* red2 = (float*)(smem + 53248);
  float* mean = (float*)(smem + 54272);
  float* rstd = (float*)(smem + 54528);
  int tid = threadIdx.x;
  int wave = tid >> 5, lane = tid & 31;
  int pos0 = blockIdx.x * 64;

  ln_tile_bf16(zsum, ln5w, ln5b, pos0, smA, red, red2, mean, rstd, tid);
  __syncthreads();
  stage_weight(l5w, smW, tid);
  __syncthreads();
  v8f acc[4] = {{}, {}, {}, {}};
  gemm_64x128(smA, smW, acc, wave, lane);
  __syncthreads();

  float* sOutF = (float*)smW;  // [ch][pos], ld 64, 32KB
  int ng = (wave & 1) * 4, m0 = (wave >> 1) * 16;
  int nl = lane & 15, mbase = (lane >> 4) * 8;
#pragma unroll
  for (int t = 0; t < 4; ++t) {
    int n = (ng + t) * 16 + nl;
    float bias = l5b[n];
#pragma unroll
    for (int r = 0; r < 8; ++r) {
      int m = m0 + mbase + r;
      float v = acc[t][r] + bias;
      v *= (float)gate5[(size_t)(pos0 + m) * 128 + n];
      v = (v > 0.f) ? v : (__expf(v) - 1.f);  // ELU(alpha=1)
      sOutF[n * 64 + m] = v;
    }
  }
  __syncthreads();
  {  // 128 channel rows x 256B coalesced f32 stores -> (C,L,L)
    int row = tid >> 1, half = tid & 1;
    const uint4* s = (const uint4*)(sOutF + row * 64 + half * 32);
    uint4* d = (uint4*)(out + (size_t)row * LL + pos0 + half * 32);
#pragma unroll
    for (int u = 0; u < 8; ++u) d[u] = s[u];
  }
}

// ------------------------------- launcher ----------------------------------
extern "C" void kernel_launch(void* const* d_in, const int* in_sizes, int n_in,
                              void* d_out, int out_size, void* d_ws,
                              size_t ws_size, hipStream_t stream) {
  (void)in_sizes; (void)n_in; (void)out_size; (void)ws_size;
  const float* z1 = (const float*)d_in[0];
  const float* z2 = (const float*)d_in[1];
  const float* z3 = (const float*)d_in[2];
  const float* ln1w = (const float*)d_in[3];
  const float* ln1b = (const float*)d_in[4];
  const float* ln2w = (const float*)d_in[5];
  const float* ln2b = (const float*)d_in[6];
  const float* ln3w = (const float*)d_in[7];
  const float* ln3b = (const float*)d_in[8];
  const float* ln5w = (const float*)d_in[9];
  const float* ln5b = (const float*)d_in[10];
  const float* l1w = (const float*)d_in[11];
  const float* l1b = (const float*)d_in[12];
  const float* l2w = (const float*)d_in[13];
  const float* l2b = (const float*)d_in[14];
  const float* l3w = (const float*)d_in[15];
  const float* l3b = (const float*)d_in[16];
  const float* l4w = (const float*)d_in[17];
  const float* l4b = (const float*)d_in[18];
  const float* g1w = (const float*)d_in[19];
  const float* g1b = (const float*)d_in[20];
  const float* g2w = (const float*)d_in[21];
  const float* g2b = (const float*)d_in[22];
  const float* g3w = (const float*)d_in[23];
  const float* g3b = (const float*)d_in[24];
  const float* g4w = (const float*)d_in[25];
  const float* g4b = (const float*)d_in[26];
  const float* l5w = (const float*)d_in[27];
  const float* l5b = (const float*)d_in[28];
  const float* g5w = (const float*)d_in[29];
  const float* g5b = (const float*)d_in[30];

  const size_t NELEM = (size_t)LL * 128;  // 33,554,432 bf16 per tensor
  bf16* z12 = (bf16*)d_ws;
  bf16* z13 = z12 + NELEM;
  bf16* p2 = z13 + NELEM;
  bf16* p3 = p2 + NELEM;
  bf16* gate5 = p3 + NELEM;
  bf16* zsum = gate5 + NELEM;

  prep_kernel<<<4096, 256, 0, stream>>>(z1, z2, z3, ln1w, ln1b, ln2w, ln2b,
                                        ln3w, ln3b, l1w, l1b, g1w, g1b, l2w,
                                        l2b, g2w, g2b, l3w, l3b, g3w, g3b, l4w,
                                        l4b, g4w, g4b, g5w, g5b, z12, z13, p2,
                                        p3, gate5);
  tri_kernel<<<dim3(64, 128), 256, 0, stream>>>(z12, z13, p2, p3, zsum);
  final_kernel<<<4096, 256, 0, stream>>>(zsum, gate5, ln5w, ln5b, l5w, l5b,
                                         (float*)d_out);
}
